// HypergraphConv_68839735820936
// MI455X (gfx1250) — matrix operand order repaired and checked by reading, not compiled, not run on hardware
//
#include <hip/hip_runtime.h>
#include <hip/hip_bf16.h>

typedef __attribute__((ext_vector_type(8)))  _Float16 v8h;
typedef __attribute__((ext_vector_type(16))) _Float16 v16h;
typedef __attribute__((ext_vector_type(8)))  float    v8f;

constexpr int B_ = 8, N_ = 4096, E_ = 2048, DIN = 256, DOUT = 256;
#define EPS_ 1e-8f

// ---------------- CDNA5 async-to-LDS helpers ----------------

// DMA 16 bytes from global memory directly into LDS (ASYNCcnt-tracked).
__device__ __forceinline__ void async_copy_b128(const _Float16* g, _Float16* l) {
  unsigned loff = (unsigned)(size_t)(__attribute__((address_space(3))) _Float16*)l;
  asm volatile("global_load_async_to_lds_b128 %0, %1, off"
               :: "v"(loff), "v"(g)
               : "memory");
}

__device__ __forceinline__ void wait_asynccnt0() {
  asm volatile("s_wait_asynccnt 0x0" ::: "memory");
}

// ---------------- prep kernels ----------------

// s[e] = sqrt(w[e] / (de[e] + eps)), de = column sums of H
__global__ void edge_scale_kernel(const float* __restrict__ H,
                                  const float* __restrict__ w,
                                  float* __restrict__ s) {
  int e = blockIdx.x * blockDim.x + threadIdx.x;
  if (e >= E_) return;
  float de = 0.f;
  for (int i = 0; i < N_; ++i) de += H[(size_t)i * E_ + e];
  s[e] = sqrtf(w[e] / (de + EPS_));
}

// dv_is[i] = (H[i,:]·w + eps)^{-1/2}; one 256-thread block per row
__global__ void node_scale_kernel(const float* __restrict__ H,
                                  const float* __restrict__ w,
                                  float* __restrict__ dv_is) {
  __shared__ float red[256];
  int i = blockIdx.x;
  float acc = 0.f;
  for (int e = threadIdx.x; e < E_; e += 256)
    acc += H[(size_t)i * E_ + e] * w[e];
  red[threadIdx.x] = acc;
  __syncthreads();
  for (int sft = 128; sft > 0; sft >>= 1) {
    if (threadIdx.x < sft) red[threadIdx.x] += red[threadIdx.x + sft];
    __syncthreads();
  }
  if (threadIdx.x == 0) dv_is[i] = 1.0f / sqrtf(red[0] + EPS_);
}

// Hs[i,e] = f16( H[i,e] * s[e] )
__global__ void build_hs_kernel(const float* __restrict__ H,
                                const float* __restrict__ s,
                                _Float16* __restrict__ Hs) {
  int e = blockIdx.x * 256 + threadIdx.x;
  int i = blockIdx.y;
  size_t idx = (size_t)i * E_ + e;
  Hs[idx] = (_Float16)(H[idx] * s[e]);
}

__global__ void cvt_f32_f16_kernel(const float* __restrict__ in,
                                   _Float16* __restrict__ out, int n) {
  int idx = blockIdx.x * 256 + threadIdx.x;
  if (idx < n) out[idx] = (_Float16)in[idx];
}

// ---------------- WMMA GEMM: C = X · Yᵀ ----------------
// X: [M x K] f16 row-major (ldx), Y: [Nn x K] f16 row-major (ldy).
// Block tile 128x128, BK=64, 256 threads = 8 waves, each wave 2x4 WMMA tiles.
// Double-buffered LDS filled by global_load_async_to_lds_b128.
// MODE 1: Ch = f16(C + bias[row]); batched over Y/C.
// MODE 2: Ch = f16(rowScale[row] * C * colScale[col]).
// MODE 3: Cf = C (f32); batched over Y/C.
template <int MODE>
__global__ __launch_bounds__(256)
void gemm_abt_wmma(const _Float16* __restrict__ X,
                   const _Float16* __restrict__ Y,
                   float* __restrict__ Cf,
                   _Float16* __restrict__ Ch,
                   const float* __restrict__ rowScale,
                   const float* __restrict__ colScale,
                   const float* __restrict__ bias,
                   int K, int ldx, int ldy, int ldc,
                   long long ybatch, long long cbatch) {
  // 64 K-halves + 8 pad per row: 144B row stride, rows stay 16B-aligned
  __shared__ _Float16 Xs[2][128][72];
  __shared__ _Float16 Ys[2][128][72];

  const int tid   = threadIdx.x;
  const int lane  = tid & 31;
  const int wave  = tid >> 5;
  const int lhalf = lane >> 4;   // 0: lanes 0-15, 1: lanes 16-31
  const int lrow  = lane & 15;
  const int wm = (wave & 3) * 32;  // wave's 32-row stripe within 128
  const int wn = (wave >> 2) * 64; // wave's 64-col stripe within 128

  const int m0 = blockIdx.y * 128;
  const int n0 = blockIdx.x * 128;

  const _Float16* Yb = Y + (size_t)blockIdx.z * ybatch;

  // Each thread DMAs 4 16B chunks per 128x64 tile: row r0+{0,32,64,96}, cols kc0..kc0+7
  const int r0  = tid >> 3;        // 0..31
  const int kc0 = (tid & 7) * 8;   // 0..56
  const _Float16* gx = X  + (size_t)(m0 + r0) * ldx + kc0;
  const _Float16* gy = Yb + (size_t)(n0 + r0) * ldy + kc0;

  auto stage = [&](int buf, int k0) {
#pragma unroll
    for (int rr = 0; rr < 128; rr += 32) {
      async_copy_b128(gx + (size_t)rr * ldx + k0, &Xs[buf][r0 + rr][kc0]);
      async_copy_b128(gy + (size_t)rr * ldy + k0, &Ys[buf][r0 + rr][kc0]);
    }
  };

  v8f acc[2][4];
#pragma unroll
  for (int a = 0; a < 2; ++a)
#pragma unroll
    for (int q = 0; q < 4; ++q) {
      v8f z = {};
      acc[a][q] = z;
    }

  stage(0, 0);
  int buf = 0;
  for (int k0 = 0; k0 < K; k0 += 64) {
    wait_asynccnt0();      // this wave's DMA into buf complete
    __syncthreads();       // all waves' DMA complete; all done reading buf^1
    if (k0 + 64 < K) stage(buf ^ 1, k0 + 64);  // overlap with compute below

#pragma unroll
    for (int kk = 0; kk < 64; kk += 32) {
      // B fragments: lane holds col N=lrow, K = kk + 16*lhalf + 0..15
      v16h bf[4];
#pragma unroll
      for (int q = 0; q < 4; ++q) {
        const v8h blo = *(const v8h*)&Ys[buf][wn + q * 16 + lrow][kk + lhalf * 16];
        const v8h bhi = *(const v8h*)&Ys[buf][wn + q * 16 + lrow][kk + lhalf * 16 + 8];
        bf[q] = __builtin_shufflevector(blo, bhi,
                0,1,2,3,4,5,6,7,8,9,10,11,12,13,14,15);
      }
      // A fragments: lane holds row M=lrow, K = kk + 16*lhalf + 0..15
#pragma unroll
      for (int a = 0; a < 2; ++a) {
        const v8h alo = *(const v8h*)&Xs[buf][wm + a * 16 + lrow][kk + lhalf * 16];
        const v8h ahi = *(const v8h*)&Xs[buf][wm + a * 16 + lrow][kk + lhalf * 16 + 8];
        v16h af = __builtin_shufflevector(alo, ahi,
                 0,1,2,3,4,5,6,7,8,9,10,11,12,13,14,15);
#pragma unroll
        for (int q = 0; q < 4; ++q) {
          acc[a][q] = __builtin_amdgcn_wmma_f32_16x16x32_f16(
              false, af, false, bf[q], (short)0, acc[a][q], false, false);
        }
      }
    }
    buf ^= 1;
  }

  // Epilogue. C layout: vgpr r -> M = tile_m + r + 8*lhalf, col N = tile_n + lrow.
#pragma unroll
  for (int a = 0; a < 2; ++a) {
#pragma unroll
    for (int q = 0; q < 4; ++q) {
      const int gn = n0 + wn + q * 16 + lrow;
#pragma unroll
      for (int r = 0; r < 8; ++r) {
        const int gm = m0 + wm + a * 16 + lhalf * 8 + r;
        float c = acc[a][q][r];
        if (MODE == 1) {
          c += bias[gm];
          (Ch + (size_t)blockIdx.z * cbatch)[(size_t)gm * ldc + gn] = (_Float16)c;
        } else if (MODE == 2) {
          c *= rowScale[gm] * colScale[gn];
          Ch[(size_t)gm * ldc + gn] = (_Float16)c;
        } else {
          (Cf + (size_t)blockIdx.z * cbatch)[(size_t)gm * ldc + gn] = c;
        }
      }
    }
  }
}

// ---------------- launch ----------------

extern "C" void kernel_launch(void* const* d_in, const int* in_sizes, int n_in,
                              void* d_out, int out_size, void* d_ws, size_t ws_size,
                              hipStream_t stream) {
  const float* x    = (const float*)d_in[0];  // [B,N,Din]
  const float* ew   = (const float*)d_in[1];  // [E]
  const float* H    = (const float*)d_in[2];  // [N,E]
  const float* W    = (const float*)d_in[3];  // [Dout,Din]
  const float* bias = (const float*)d_in[4];  // [Dout]
  float* out = (float*)d_out;                 // [B,N,Dout]

  auto align256 = [](size_t v) { return (v + 255) & ~(size_t)255; };
  char* ws = (char*)d_ws;
  float* s      = (float*)ws;     ws += align256((size_t)E_ * 4);
  float* dv_is  = (float*)ws;     ws += align256((size_t)N_ * 4);
  _Float16* Wh  = (_Float16*)ws;  ws += align256((size_t)DOUT * DIN * 2);
  _Float16* xh  = (_Float16*)ws;  ws += align256((size_t)B_ * N_ * DIN * 2);
  _Float16* xwT = (_Float16*)ws;  ws += align256((size_t)B_ * DOUT * N_ * 2);
  _Float16* Hs  = (_Float16*)ws;  ws += align256((size_t)N_ * E_ * 2);
  _Float16* Ah  = (_Float16*)ws;  // [N,N] f16

  edge_scale_kernel<<<(E_ + 255) / 256, 256, 0, stream>>>(H, ew, s);
  node_scale_kernel<<<N_, 256, 0, stream>>>(H, ew, dv_is);
  build_hs_kernel<<<dim3(E_ / 256, N_), 256, 0, stream>>>(H, s, Hs);
  cvt_f32_f16_kernel<<<(DOUT * DIN + 255) / 256, 256, 0, stream>>>(W, Wh, DOUT * DIN);
  cvt_f32_f16_kernel<<<(B_ * N_ * DIN + 255) / 256, 256, 0, stream>>>(x, xh, B_ * N_ * DIN);

  // xwT[b] = W · x[b]ᵀ + bias   (M=DOUT, Nn=N, K=DIN) -> f16 [B][DOUT][N]
  gemm_abt_wmma<1><<<dim3(N_ / 128, DOUT / 128, B_), 256, 0, stream>>>(
      Wh, xh, nullptr, xwT, nullptr, nullptr, bias,
      DIN, DIN, DIN, N_, (long long)N_ * DIN, (long long)DOUT * N_);

  // Ah = dv_is ⊙ (Hs · Hsᵀ) ⊙ dv_is   (M=Nn=N, K=E) -> f16 [N][N]
  gemm_abt_wmma<2><<<dim3(N_ / 128, N_ / 128, 1), 256, 0, stream>>>(
      Hs, Hs, nullptr, Ah, dv_is, dv_is, nullptr,
      E_, E_, E_, N_, 0, 0);

  // out[b] = Ah · xw[b] = Ah · (xwT[b])ᵀ   (M=N, Nn=DOUT, K=N) -> f32
  gemm_abt_wmma<3><<<dim3(DOUT / 128, N_ / 128, B_), 256, 0, stream>>>(
      Ah, xwT, out, nullptr, nullptr, nullptr, nullptr,
      N_, N_, N_, DOUT, (long long)DOUT * N_, (long long)N_ * DOUT);
}
